// Sparse_Spatial_Attention_7584912244771
// MI455X (gfx1250) — matrix operand order repaired and checked by reading, not compiled, not run on hardware
//
#include <hip/hip_runtime.h>

// ---- problem constants (from reference) ----
#define BB     8
#define TT     12
#define NN     512
#define DD     64
#define HH     8
#define DHD    8
#define KADJ_  16
#define SNODES 27              // int(3 * log2(512))
#define M64    (BB * TT * NN)  // 49152 rows of the D=64 GEMMs

typedef __attribute__((ext_vector_type(16))) __bf16 v16bf;
typedef __attribute__((ext_vector_type(8)))  __bf16 v8bf;
typedef __attribute__((ext_vector_type(8)))  float  v8f;
typedef __attribute__((ext_vector_type(2)))  float  v2f;

// -------------------------------------------------------------------------
// Kernel 1: x_ = x + spa_eigvec*spa_eigvalue + tem_eigvec*tem_eigvalue, to bf16
// -------------------------------------------------------------------------
__global__ void prep_x_kernel(const float* __restrict__ x,
                              const float* __restrict__ spa_val,
                              const float* __restrict__ spa_vec,
                              const float* __restrict__ tem_val,
                              const float* __restrict__ tem_vec,
                              __bf16* __restrict__ xb) {
  int idx = blockIdx.x * blockDim.x + threadIdx.x;
  if (idx >= M64 * DD) return;
  int d = idx & (DD - 1);
  int n = (idx >> 6) & (NN - 1);
  float v = x[idx] + spa_vec[n * DD + d] * spa_val[d]
                   + tem_vec[n * DD + d] * tem_val[d];
  xb[idx] = (__bf16)v;
}

// -------------------------------------------------------------------------
// Kernel 2: convert the five D x D weight matrices to bf16 (one packed buffer)
// -------------------------------------------------------------------------
__global__ void conv_w_kernel(const float* __restrict__ w0, const float* __restrict__ w1,
                              const float* __restrict__ w2, const float* __restrict__ w3,
                              const float* __restrict__ w4, __bf16* __restrict__ out) {
  int idx = blockIdx.x * blockDim.x + threadIdx.x;
  if (idx >= 5 * DD * DD) return;
  int sel = idx >> 12;          // DD*DD == 4096
  int j   = idx & 4095;
  const float* src = (sel == 0) ? w0 : (sel == 1) ? w1 : (sel == 2) ? w2
                   : (sel == 3) ? w3 : w4;
  out[idx] = (__bf16)src[j];
}

// -------------------------------------------------------------------------
// Kernel 3: generic out[m,j] = sum_d A[m,d]*W[j,d] + bias[j]   (M64 x 64, K=64)
// bf16 WMMA, f32 accumulate. Block = 256 threads (8 waves); each wave owns a
// full 16x64 output strip (one m-tile): A loaded once per k-step, 4 WMMAs
// against the 4 weight n-tiles -> 8 WMMAs per wave, 4x less redundant A
// traffic than one-tile-per-wave. K=64 -> two k=32 WMMA steps.
// -------------------------------------------------------------------------
__global__ void __launch_bounds__(256)
gemm_d64_kernel(const __bf16* __restrict__ A, const __bf16* __restrict__ W,
                const float* __restrict__ bias, float* __restrict__ out) {
  const int lane  = threadIdx.x & 31;
  const int wave  = threadIdx.x >> 5;              // m-tile within block
  const int mtile = blockIdx.x * 8 + wave;         // 0..M64/16-1
  const int lmod  = lane & 15;
  const int hi    = lane >> 4;                     // lane half
  const int arow  = mtile * 16 + lmod;             // A-matrix row (M)

  // prefetch the A rows the next block will consume (gfx1250 global_prefetch)
  if (arow + 128 < M64)
    __builtin_prefetch(A + (size_t)(arow + 128) * DD, 0, 1);

  v8f c0 = {}, c1 = {}, c2 = {}, c3 = {};
#pragma unroll
  for (int k0 = 0; k0 < DD; k0 += 32) {
    // A 16x32 bf16, contiguous k order used here (B matches): lane<16 holds K 0..15; lane>=16 K 16..31
    int abase = k0 + hi * 16;
    v8bf alo = *(const v8bf*)(A + (size_t)arow * DD + abase);
    v8bf ahi = *(const v8bf*)(A + (size_t)arow * DD + abase + 8);
    v16bf av = __builtin_shufflevector(alo, ahi,
        0,1,2,3,4,5,6,7,8,9,10,11,12,13,14,15);
    // B 32x16 bf16, same contiguous k order as A: lane<16 holds K 0..15 of column lmod, lane>=16 K 16..31
    // B[d][j] = W[j][d]  =>  contiguous row of W
    int d0 = k0 + hi * 16;
#pragma unroll
    for (int nt = 0; nt < 4; ++nt) {
      int jcol = nt * 16 + lmod;
      v8bf blo = *(const v8bf*)(W + jcol * DD + d0);
      v8bf bhi = *(const v8bf*)(W + jcol * DD + d0 + 8);
      v16bf bvv = __builtin_shufflevector(blo, bhi,
          0,1,2,3,4,5,6,7,8,9,10,11,12,13,14,15);
      v8f cin = (nt == 0) ? c0 : (nt == 1) ? c1 : (nt == 2) ? c2 : c3;
      v8f cot = __builtin_amdgcn_wmma_f32_16x16x32_bf16(
          false, av, false, bvv, (short)0, cin, false, false);
      if (nt == 0) c0 = cot; else if (nt == 1) c1 = cot;
      else if (nt == 2) c2 = cot; else c3 = cot;
    }
  }
#pragma unroll
  for (int nt = 0; nt < 4; ++nt) {
    int jcol = nt * 16 + lmod;
    float bj = bias[jcol];
    v8f c = (nt == 0) ? c0 : (nt == 1) ? c1 : (nt == 2) ? c2 : c3;
#pragma unroll
    for (int r = 0; r < 8; ++r) {
      int mrow = mtile * 16 + r + hi * 8;          // C layout: M = r + 8*(lane>=16)
      out[(size_t)mrow * DD + jcol] = c[r] + bj;
    }
  }
}

// -------------------------------------------------------------------------
// Kernel 4: sparse spatial attention core. One workgroup (256 thr = 8 waves)
// per (bh, t). Stages: sampled scores -> top-27 -> f32 WMMA QK^T + masks ->
// softmax -> argmax over S -> attn@V -> gather into merged bf16.
// -------------------------------------------------------------------------
__global__ void __launch_bounds__(256)
sparse_attn_kernel(const float* __restrict__ Qf, const float* __restrict__ Kf,
                   const float* __restrict__ Vf,
                   const float* __restrict__ mask1, const float* __restrict__ mask2,
                   const float* __restrict__ proj_w, const float* __restrict__ proj_b,
                   const int* __restrict__ localadj,
                   const int* __restrict__ isM1p, const int* __restrict__ isM2p,
                   __bf16* __restrict__ mergedb) {
  __shared__ float sc[28][NN];        // 57344 B: scores / attn (rows 0..26 valid)
  __shared__ int   mtop[32];
  __shared__ float qr[32][DHD];       // Q_reduce, zero-padded to 32 rows
  __shared__ float valbuf[28][DHD];
  __shared__ float wred_v[8];
  __shared__ int   wred_i[8];
  __shared__ int   gsel_sh;
  float* Mbuf = &sc[0][0];            // alias: M scores live here before sc is used

  const int tid  = threadIdx.x;
  const int lane = tid & 31;
  const int wave = tid >> 5;
  const int gid  = blockIdx.x;
  const int bh   = gid / TT;
  const int t    = gid % TT;
  const int h    = bh / BB;           // head_ids = repeat(arange(H), B)
  const int b    = bh % BB;
  const size_t plane = ((size_t)(b * TT + t)) * NN * DD;
  const int hoff = h * DHD;
  const int isM1 = isM1p[0], isM2 = isM2p[0];
  const float pb = proj_b[0];

  // ---- Stage 1: M[n] = sum_k dot(Q[n], K[adj[n,k]]) * proj_w[k] + proj_b ----
  for (int n = tid; n < NN; n += 256) {
    const float* qrow = Qf + plane + (size_t)n * DD + hoff;
    float q[DHD];
#pragma unroll
    for (int kk = 0; kk < DHD; ++kk) q[kk] = qrow[kk];
    float acc = pb;
#pragma unroll
    for (int k = 0; k < KADJ_; ++k) {
      int a = localadj[n * KADJ_ + k];
      const float* krow = Kf + plane + (size_t)a * DD + hoff;
      float dot = 0.f;
#pragma unroll
      for (int kk = 0; kk < DHD; ++kk) dot += q[kk] * krow[kk];
      acc += dot * proj_w[k];
    }
    Mbuf[n] = acc;
  }
  __syncthreads();

  // ---- Stage 2: iterative top-27 (descending, first-index tie-break) ----
  float v0 = Mbuf[tid];
  float v1 = Mbuf[tid + 256];
  for (int s = 0; s < SNODES; ++s) {
    float bvv; int bii;
    if (v0 >= v1) { bvv = v0; bii = tid; } else { bvv = v1; bii = tid + 256; }
#pragma unroll
    for (int m = 16; m >= 1; m >>= 1) {
      float ov = __shfl_xor(bvv, m);
      int   oi = __shfl_xor(bii, m);
      if (ov > bvv || (ov == bvv && oi < bii)) { bvv = ov; bii = oi; }
    }
    if (lane == 0) { wred_v[wave] = bvv; wred_i[wave] = bii; }
    __syncthreads();
    if (tid == 0) {
      float gv = wred_v[0]; int gi = wred_i[0];
      for (int w2 = 1; w2 < 8; ++w2)
        if (wred_v[w2] > gv || (wred_v[w2] == gv && wred_i[w2] < gi)) {
          gv = wred_v[w2]; gi = wred_i[w2];
        }
      mtop[s] = gi; gsel_sh = gi;
    }
    __syncthreads();
    int gi = gsel_sh;
    if (gi == tid)       v0 = -3.0e38f;
    if (gi == tid + 256) v1 = -3.0e38f;
  }

  // ---- Stage 3: gather Q_reduce (zero-pad rows 27..31) ----
  {
    int s = tid >> 3, d = tid & 7;    // 256 = 32 x 8
    float qv = 0.f;
    if (s < SNODES) qv = Qf[plane + (size_t)mtop[s] * DD + hoff + d];
    qr[s][d] = qv;
  }
  __syncthreads();

  // ---- Stage 4: Q_K = (Q_reduce @ K^T)/sqrt(8), masks; f32 WMMA 16x16x4 ----
  // 64 output tiles: 2 s-tiles x 32 n-tiles, round-robin over 8 waves.
  const float scale = 0.35355339059327373f;  // 1/sqrt(DH)
  for (int tt2 = wave; tt2 < 64; tt2 += 8) {
    int s_tile = tt2 & 1;
    int n_tile = tt2 >> 1;
    int lmod = lane & 15;
    int kk2  = (lane >> 4) << 1;      // 0 or 2
    v8f c = {};
#pragma unroll
    for (int k0 = 0; k0 < DHD; k0 += 4) {
      int kk = k0 + kk2;
      v2f av, bvv;
      int srow = s_tile * 16 + lmod;
      av[0] = qr[srow][kk];  av[1] = qr[srow][kk + 1];     // A 16x4: V0=K kk, V1=K kk+1
      int ncol = n_tile * 16 + lmod;
      const float* krow = Kf + plane + (size_t)ncol * DD + hoff + kk;
      bvv[0] = krow[0];      bvv[1] = krow[1];             // B 4x16: K^T[k][n] = K[n][k]
      c = __builtin_amdgcn_wmma_f32_16x16x4_f32(
          false, av, false, bvv, (short)0, c, false, false);
    }
    int ncol = n_tile * 16 + lmod;
#pragma unroll
    for (int r = 0; r < 8; ++r) {
      int s = s_tile * 16 + r + ((lane >> 4) << 3);
      if (s < 28) {
        float v = c[r] * scale;
        if (s < SNODES) {
          int node = mtop[s];
          if (isM1) v *= mask1[(size_t)node * NN + ncol];
          if (isM2) v *= mask2[((size_t)h * NN + node) * NN + ncol];
        } else {
          v = 0.f;
        }
        sc[s][ncol] = v;
      }
    }
  }
  __syncthreads();

  // ---- Stage 5: softmax over N per valid row s (wave-parallel) ----
  for (int s = wave; s < SNODES; s += 8) {
    float m = -3.0e38f;
    for (int n = lane; n < NN; n += 32) m = fmaxf(m, sc[s][n]);
#pragma unroll
    for (int mm = 16; mm >= 1; mm >>= 1) m = fmaxf(m, __shfl_xor(m, mm));
    float ssum = 0.f;
    for (int n = lane; n < NN; n += 32) {
      float e = __expf(sc[s][n] - m);
      sc[s][n] = e;
      ssum += e;
    }
#pragma unroll
    for (int mm = 16; mm >= 1; mm >>= 1) ssum += __shfl_xor(ssum, mm);
    float inv = 1.0f / ssum;
    for (int n = lane; n < NN; n += 32) sc[s][n] *= inv;
  }
  __syncthreads();

  // ---- Stage 6: cp[n] = argmax_s attn[s][n] (kept in registers) ----
  int cp0 = 0, cp1 = 0;
  {
    float b0 = sc[0][tid], b1 = sc[0][tid + 256];
    for (int s = 1; s < SNODES; ++s) {
      float x0 = sc[s][tid];       if (x0 > b0) { b0 = x0; cp0 = s; }
      float x1 = sc[s][tid + 256]; if (x1 > b1) { b1 = x1; cp1 = s; }
    }
  }

  // ---- Stage 7: val[s][d] = sum_n attn[s][n] * V[n][d] ----
  if (tid < SNODES * DHD) {
    int s = tid >> 3, d = tid & 7;
    const float* vbase = Vf + plane + hoff + d;
    float acc = 0.f;
    for (int n = 0; n < NN; ++n) acc += sc[s][n] * vbase[(size_t)n * DD];
    valbuf[s][d] = acc;
  }
  __syncthreads();

  // ---- Stage 8: out_v[n] = val[cp[n]]; write merged-heads bf16 ----
  {
    size_t obase = plane + hoff;
#pragma unroll
    for (int d = 0; d < DHD; ++d)
      mergedb[obase + (size_t)tid * DD + d] = (__bf16)valbuf[cp0][d];
#pragma unroll
    for (int d = 0; d < DHD; ++d)
      mergedb[obase + (size_t)(tid + 256) * DD + d] = (__bf16)valbuf[cp1][d];
  }
}

// -------------------------------------------------------------------------
// Kernel 5: per-row LayerNorm (D=64), f32 in -> bf16 out (feeds Wff GEMM)
// -------------------------------------------------------------------------
__global__ void layernorm_kernel(const float* __restrict__ vin,
                                 const float* __restrict__ g,
                                 const float* __restrict__ be,
                                 __bf16* __restrict__ outb) {
  int row = blockIdx.x * blockDim.x + threadIdx.x;
  if (row >= M64) return;
  const float* p = vin + (size_t)row * DD;
  float s = 0.f, s2 = 0.f;
  for (int d = 0; d < DD; ++d) { float v = p[d]; s += v; s2 += v * v; }
  float mu  = s * (1.0f / DD);
  float var = s2 * (1.0f / DD) - mu * mu;
  float r   = rsqrtf(var + 1e-5f);
  __bf16* o = outb + (size_t)row * DD;
  for (int d = 0; d < DD; ++d)
    o[d] = (__bf16)((p[d] - mu) * r * g[d] + be[d]);
}

// -------------------------------------------------------------------------
// Host launcher
// -------------------------------------------------------------------------
extern "C" void kernel_launch(void* const* d_in, const int* in_sizes, int n_in,
                              void* d_out, int out_size, void* d_ws, size_t ws_size,
                              hipStream_t stream) {
  (void)in_sizes; (void)n_in; (void)out_size; (void)ws_size;
  const float* x        = (const float*)d_in[0];
  const float* spa_val  = (const float*)d_in[1];
  const float* spa_vec  = (const float*)d_in[2];
  const float* tem_val  = (const float*)d_in[3];
  const float* tem_vec  = (const float*)d_in[4];
  const float* mask1    = (const float*)d_in[5];
  const float* mask2    = (const float*)d_in[6];
  const float* Wq       = (const float*)d_in[7];
  const float* bq       = (const float*)d_in[8];
  const float* Wk       = (const float*)d_in[9];
  const float* bk       = (const float*)d_in[10];
  const float* Wv       = (const float*)d_in[11];
  const float* bv       = (const float*)d_in[12];
  const float* Wo       = (const float*)d_in[13];
  const float* bo       = (const float*)d_in[14];
  const float* Wff      = (const float*)d_in[15];
  const float* bff      = (const float*)d_in[16];
  const float* ln_g     = (const float*)d_in[17];
  const float* ln_b     = (const float*)d_in[18];
  const float* proj_w   = (const float*)d_in[19];
  const float* proj_b   = (const float*)d_in[20];
  const int*   localadj = (const int*)d_in[21];
  const int*   isM1     = (const int*)d_in[22];
  const int*   isM2     = (const int*)d_in[23];

  // workspace carving (256-byte aligned regions)
  char* wsb = (char*)d_ws;
  size_t off = 0;
  auto take = [&](size_t bytes) -> char* {
    char* p = wsb + off;
    off += bytes;
    off = (off + 255) & ~(size_t)255;
    return p;
  };
  __bf16* xb  = (__bf16*)take((size_t)M64 * DD * 2);   // x_ in bf16
  __bf16* wb  = (__bf16*)take((size_t)5 * DD * DD * 2);// Wq,Wk,Wv,Wo,Wff bf16
  float*  Qf  = (float*)take((size_t)M64 * DD * 4);
  float*  Kf  = (float*)take((size_t)M64 * DD * 4);
  float*  Vf  = (float*)take((size_t)M64 * DD * 4);
  __bf16* mgb = (__bf16*)take((size_t)M64 * DD * 2);   // merged heads bf16
  float*  vof = (float*)take((size_t)M64 * DD * 4);    // after Wo
  __bf16* lnb = (__bf16*)take((size_t)M64 * DD * 2);   // after LN, bf16

  const dim3 blk256(256);
  const dim3 gemm_grid(M64 / 16 / 8);                  // 384 blocks, 8 waves each

  prep_x_kernel<<<dim3((M64 * DD + 255) / 256), blk256, 0, stream>>>(
      x, spa_val, spa_vec, tem_val, tem_vec, xb);
  conv_w_kernel<<<dim3((5 * DD * DD + 255) / 256), blk256, 0, stream>>>(
      Wq, Wk, Wv, Wo, Wff, wb);

  gemm_d64_kernel<<<gemm_grid, blk256, 0, stream>>>(xb, wb + 0 * DD * DD, bq, Qf);
  gemm_d64_kernel<<<gemm_grid, blk256, 0, stream>>>(xb, wb + 1 * DD * DD, bk, Kf);
  gemm_d64_kernel<<<gemm_grid, blk256, 0, stream>>>(xb, wb + 2 * DD * DD, bv, Vf);

  sparse_attn_kernel<<<dim3(HH * BB * TT), blk256, 0, stream>>>(
      Qf, Kf, Vf, mask1, mask2, proj_w, proj_b, localadj, isM1, isM2, mgb);

  gemm_d64_kernel<<<gemm_grid, blk256, 0, stream>>>(mgb, wb + 3 * DD * DD, bo, vof);
  layernorm_kernel<<<dim3((M64 + 255) / 256), blk256, 0, stream>>>(vof, ln_g, ln_b, lnb);
  gemm_d64_kernel<<<gemm_grid, blk256, 0, stream>>>(lnb, wb + 4 * DD * DD, bff,
                                                    (float*)d_out);
}
